// GraphQNNGen462_65481071403176
// MI455X (gfx1250) — matrix operand order repaired and checked by reading, not compiled, not run on hardware
//
#include <hip/hip_runtime.h>
#include <hip/hip_bf16.h>

// ---------------------------------------------------------------------------
// GCN (3 layers, 128ch) + MLP head (128->512->256), N=50000 nodes, E=800000.
//   h' = (x @ W) * dinv[row]            (WMMA f16->f32 GEMM, epilogue scale)
//   agg[d] += h'[s]  over edges         (f16 gather, f32 global atomics)
//   x_next = relu(dinv*(agg + h') + b)  (self-loop folded in, f16 out)
//   MLP: two WMMA GEMMs with bias+relu epilogues (last writes f32 d_out).
// B weights pre-repacked as [ks][nt][lane][16 halves] so the GEMM k-loop is
// pure pointer-increment + immediate-offset b128 loads feeding v_wmma.
// ---------------------------------------------------------------------------

typedef _Float16 f16;
typedef __attribute__((ext_vector_type(16))) _Float16 v16h;
typedef __attribute__((ext_vector_type(4)))  _Float16 v4h;
typedef __attribute__((ext_vector_type(8)))  float    v8f;

#define NNODES 50000
#define NEDGES 800000
#define HID    128

// ---------------- small utility kernels ----------------

__global__ void k_fill1(float* __restrict__ p, int n) {
    int t = blockIdx.x * 256 + threadIdx.x;
    if (t < n) p[t] = 1.0f;
}

__global__ void k_deg(const int* __restrict__ dst, float* __restrict__ deg, int n) {
    int t = blockIdx.x * 256 + threadIdx.x;
    if (t < n) atomicAdd(&deg[dst[t]], 1.0f);
}

__global__ void k_rsqrt(float* __restrict__ p, int n) {
    int t = blockIdx.x * 256 + threadIdx.x;
    if (t < n) p[t] = rsqrtf(p[t]);
}

__global__ void k_zero4(float4* __restrict__ p, int n4) {
    int t = blockIdx.x * 256 + threadIdx.x;
    if (t < n4) p[t] = make_float4(0.f, 0.f, 0.f, 0.f);
}

__global__ void k_f32_to_f16(const float* __restrict__ x, f16* __restrict__ y, int n) {
    int t = blockIdx.x * 256 + threadIdx.x;
    if (t < n) y[t] = (f16)x[t];
}

// Repack row-major f32 W[K][Cout] into WMMA B fragments, layout [ks][nt][lane][16h].
// Fragment (ks, nt) covers K rows ks*32..+31, N cols nt*16..+15.
// Lane L holds 16 halves: K = ks*32 + (L>=16 ? 8:0) + (h&7) + (h>=8 ? 16:0), N = nt*16 + (L&15).
__global__ void k_repack(const float* __restrict__ W, f16* __restrict__ Bp, int K, int Cout) {
    int t = blockIdx.x * 256 + threadIdx.x;        // total threads = K*Cout
    int h     = t & 15;
    int L     = (t >> 4) & 31;
    int frag  = t >> 9;                            // = ks*NTtot + nt
    int NTtot = Cout >> 4;
    int nt    = frag % NTtot;
    int ks    = frag / NTtot;
    int k     = (ks << 5) + ((L >> 4) << 3) + (h & 7) + ((h >> 3) << 4);
    int n     = (nt << 4) + (L & 15);
    Bp[t] = (f16)W[(size_t)k * Cout + n];
}

// ---------------- WMMA GEMM: C[M,Cout] = A[M,K] @ B, per-wave 16x64 strip ----------------
// EPI 0: out_f16 = acc * rowscale[row]            (GCN pre-aggregation, h')
// EPI 1: out_f16 = relu(acc + bias[col])          (MLP hidden)
// EPI 2: out_f32 = relu(acc + bias[col])          (final output)
template <int EPI>
__global__ __launch_bounds__(256) void k_wmma_gemm(
    const f16* __restrict__ A, const f16* __restrict__ Bp,
    const float* __restrict__ rowscale, const float* __restrict__ bias,
    f16* __restrict__ OutH, float* __restrict__ OutF,
    int M, int K, int Cout)
{
    const int KS    = K >> 5;            // k-steps of 32
    const int NTtot = Cout >> 4;         // 16-col tiles total
    const int NQ    = Cout >> 6;         // 64-col quads per tile-row
    const int wv    = blockIdx.x * 8 + (threadIdx.x >> 5);
    const int lane  = threadIdx.x & 31;
    const int total = (M >> 4) * NQ;
    if (wv >= total) return;
    const int mt = wv / NQ;
    const int nq = wv - mt * NQ;

    // A fragment cursor (16-bit A 16x32 ISA layout); advances 32 halves per k-step.
    const int mrow = (mt << 4) + (lane & 15);
    const int kin  = (lane >> 4) << 3;            // 0 or 8
    const f16* __restrict__ Acur = A + (size_t)mrow * K + kin;
    // B fragment cursor: 4 contiguous fragments (4 KB) per k-step, fixed lane offset.
    const f16* __restrict__ Bcur = Bp + (((size_t)(nq << 2) << 5) + lane) * 16;
    const size_t bstep = (size_t)NTtot << 9;      // NTtot*512 halves per k-step

    v8f a0 = {}, a1 = {}, a2 = {}, a3 = {};
    for (int ks = 0; ks < KS; ++ks) {
        union { v16h v; uint4 u[2]; } af, b0, b1, b2, b3;
        af.u[0] = *(const uint4*)(Acur);           // K = kin .. kin+7
        af.u[1] = *(const uint4*)(Acur + 16);      // K = kin+16 .. kin+23
        b0.u[0] = *(const uint4*)(Bcur);
        b0.u[1] = *(const uint4*)(Bcur + 8);
        b1.u[0] = *(const uint4*)(Bcur + 512);
        b1.u[1] = *(const uint4*)(Bcur + 520);
        b2.u[0] = *(const uint4*)(Bcur + 1024);
        b2.u[1] = *(const uint4*)(Bcur + 1032);
        b3.u[0] = *(const uint4*)(Bcur + 1536);
        b3.u[1] = *(const uint4*)(Bcur + 1544);
        a0 = __builtin_amdgcn_wmma_f32_16x16x32_f16(false, af.v, false, b0.v, (short)0, a0, false, false);
        a1 = __builtin_amdgcn_wmma_f32_16x16x32_f16(false, af.v, false, b1.v, (short)0, a1, false, false);
        a2 = __builtin_amdgcn_wmma_f32_16x16x32_f16(false, af.v, false, b2.v, (short)0, a2, false, false);
        a3 = __builtin_amdgcn_wmma_f32_16x16x32_f16(false, af.v, false, b3.v, (short)0, a3, false, false);
        Acur += 32;
        Bcur += bstep;
    }

    // C/D layout: VGPR r -> row mt*16 + r + (lane>=16 ? 8:0), col = tile*16 + (lane&15)
    const int rbase = (mt << 4) + ((lane >> 4) << 3);
    const int cbase = (nq << 6) + (lane & 15);
    float bv0 = 0.f, bv1 = 0.f, bv2 = 0.f, bv3 = 0.f;
    if (EPI != 0) {
        bv0 = bias[cbase];      bv1 = bias[cbase + 16];
        bv2 = bias[cbase + 32]; bv3 = bias[cbase + 48];
    }
#pragma unroll
    for (int r = 0; r < 8; ++r) {
        const int row = rbase + r;
        const size_t ro = (size_t)row * Cout;
        if (EPI == 0) {
            const float s = rowscale[row];
            OutH[ro + cbase]      = (f16)(a0[r] * s);
            OutH[ro + cbase + 16] = (f16)(a1[r] * s);
            OutH[ro + cbase + 32] = (f16)(a2[r] * s);
            OutH[ro + cbase + 48] = (f16)(a3[r] * s);
        } else if (EPI == 1) {
            OutH[ro + cbase]      = (f16)fmaxf(a0[r] + bv0, 0.f);
            OutH[ro + cbase + 16] = (f16)fmaxf(a1[r] + bv1, 0.f);
            OutH[ro + cbase + 32] = (f16)fmaxf(a2[r] + bv2, 0.f);
            OutH[ro + cbase + 48] = (f16)fmaxf(a3[r] + bv3, 0.f);
        } else {
            OutF[ro + cbase]      = fmaxf(a0[r] + bv0, 0.f);
            OutF[ro + cbase + 16] = fmaxf(a1[r] + bv1, 0.f);
            OutF[ro + cbase + 32] = fmaxf(a2[r] + bv2, 0.f);
            OutF[ro + cbase + 48] = fmaxf(a3[r] + bv3, 0.f);
        }
    }
}

// ---------------- edge scatter: one wave32 per edge, 4 channels / lane ----------------
__global__ __launch_bounds__(256) void k_scatter(
    const int* __restrict__ srcs, const int* __restrict__ dsts,
    const f16* __restrict__ hp, float* __restrict__ agg, int E)
{
    const int wv   = blockIdx.x * 8 + (threadIdx.x >> 5);
    const int lane = threadIdx.x & 31;
    if (wv >= E) return;
    const int s = srcs[wv];
    const int d = dsts[wv];
    if (wv + 32 < E) {  // prefetch a future source row (global_prefetch_b8)
        const int sf = srcs[wv + 32];
        __builtin_prefetch(hp + (size_t)sf * HID + lane * 4, 0, 0);
    }
    const v4h v = *(const v4h*)(hp + (size_t)s * HID + lane * 4);
    float* o = agg + (size_t)d * HID + lane * 4;
    atomicAdd(o + 0, (float)v[0]);
    atomicAdd(o + 1, (float)v[1]);
    atomicAdd(o + 2, (float)v[2]);
    atomicAdd(o + 3, (float)v[3]);
}

// ---------------- finalize: fold self-loop, dinv, bias, relu, cvt f16 ----------------
__global__ void k_finalize(const float* __restrict__ agg, const f16* __restrict__ hp,
                           const float* __restrict__ dinv, const float* __restrict__ bias,
                           f16* __restrict__ actOut)
{
    const int t = blockIdx.x * 256 + threadIdx.x;     // grid == N*HID/256 exactly
    const int i = t >> 7;
    const int c = t & 127;
    const float v = dinv[i] * (agg[t] + (float)hp[t]) + bias[c];
    actOut[t] = (f16)fmaxf(v, 0.f);
}

// ---------------------------------------------------------------------------

extern "C" void kernel_launch(void* const* d_in, const int* in_sizes, int n_in,
                              void* d_out, int out_size, void* d_ws, size_t ws_size,
                              hipStream_t stream)
{
    (void)in_sizes; (void)n_in; (void)out_size; (void)ws_size;
    const float* x   = (const float*)d_in[0];
    const int*   ei  = (const int*)d_in[1];            // [2, E] (src row, dst row)
    const float* Wg[3] = {(const float*)d_in[2], (const float*)d_in[4], (const float*)d_in[6]};
    const float* bg[3] = {(const float*)d_in[3], (const float*)d_in[5], (const float*)d_in[7]};
    const float* Wm0 = (const float*)d_in[8];
    const float* bm0 = (const float*)d_in[9];
    const float* Wm1 = (const float*)d_in[10];
    const float* bm1 = (const float*)d_in[11];

    // ---- workspace carve-out (~103 MB) ----
    char* ws = (char*)d_ws;
    size_t off = 0;
    auto take = [&](size_t bytes) -> char* {
        off = (off + 255) & ~(size_t)255;
        char* p = ws + off;
        off += bytes;
        return p;
    };
    float* dinv = (float*)take((size_t)NNODES * 4);
    float* agg  = (float*)take((size_t)NNODES * HID * 4);
    f16*   hp   = (f16*)  take((size_t)NNODES * HID * 2);
    f16*   actA = (f16*)  take((size_t)NNODES * HID * 2);     // GCN activations (f16)
    f16*   actB = (f16*)  take((size_t)NNODES * 512 * 2);     // MLP hidden (f16)
    f16*   BpG[3];
    for (int l = 0; l < 3; ++l) BpG[l] = (f16*)take((size_t)128 * 128 * 2);
    f16*   BpM0 = (f16*)take((size_t)128 * 512 * 2);
    f16*   BpM1 = (f16*)take((size_t)512 * 256 * 2);

    const int nh  = NNODES * HID;                 // 6,400,000 (divisible by 256 and 1024)
    const int blkNH  = nh / 256;                  // 25000
    const int blkN   = (NNODES + 255) / 256;
    const int blkE   = NEDGES / 256;              // 3125
    const int blkEd  = (NEDGES + 7) / 8;          // 100000 (one wave per edge)
    auto gemmBlocks = [](int M, int Cout) { int w = (M >> 4) * (Cout >> 6); return (w + 7) / 8; };

    // ---- weight repack into WMMA B-fragment layout ----
    for (int l = 0; l < 3; ++l)
        k_repack<<<(128 * 128) / 256, 256, 0, stream>>>(Wg[l], BpG[l], 128, 128);
    k_repack<<<(128 * 512) / 256, 256, 0, stream>>>(Wm0, BpM0, 128, 512);
    k_repack<<<(512 * 256) / 256, 256, 0, stream>>>(Wm1, BpM1, 512, 256);

    // ---- input f32 -> f16 ----
    k_f32_to_f16<<<blkNH, 256, 0, stream>>>(x, actA, nh);

    // ---- degree (self-loop => start at 1) and dinv = rsqrt(deg) ----
    k_fill1<<<blkN, 256, 0, stream>>>(dinv, NNODES);
    k_deg<<<blkE, 256, 0, stream>>>(ei + NEDGES, dinv, NEDGES);
    k_rsqrt<<<blkN, 256, 0, stream>>>(dinv, NNODES);

    // ---- 3 GCN layers ----
    for (int l = 0; l < 3; ++l) {
        k_wmma_gemm<0><<<gemmBlocks(NNODES, HID), 256, 0, stream>>>(
            actA, BpG[l], dinv, dinv, hp, nullptr, NNODES, HID, HID);
        k_zero4<<<(nh / 4) / 256, 256, 0, stream>>>((float4*)agg, nh / 4);
        k_scatter<<<blkEd, 256, 0, stream>>>(ei, ei + NEDGES, hp, agg, NEDGES);
        k_finalize<<<blkNH, 256, 0, stream>>>(agg, hp, dinv, bg[l], actA);
    }

    // ---- MLP head ----
    k_wmma_gemm<1><<<gemmBlocks(NNODES, 512), 256, 0, stream>>>(
        actA, BpM0, dinv, bm0, actB, nullptr, NNODES, 128, 512);
    k_wmma_gemm<2><<<gemmBlocks(NNODES, 256), 256, 0, stream>>>(
        actB, BpM1, dinv, bm1, nullptr, (float*)d_out, NNODES, 512, 256);
}
